// LinearAttention_2482491097603
// MI455X (gfx1250) — matrix operand order repaired and checked by reading, not compiled, not run on hardware
//
#include <hip/hip_runtime.h>

#define BATCH  4
#define SEQT   4096
#define DMODEL 1024
#define DHEAD  128
#define NPART  8
#define TPART  (SEQT / NPART)   // 512
#define CHUNK  64

typedef __bf16 bf16;
typedef bf16  v16bf __attribute__((ext_vector_type(16)));
typedef float v8f   __attribute__((ext_vector_type(8)));

// Native conversions (hardware cvt on gfx1250, replaces manual RNE bit math)
static __device__ __forceinline__ unsigned short f2bf(float f) {
  return __builtin_bit_cast(unsigned short, (__bf16)f);
}
static __device__ __forceinline__ float bf2f(unsigned short s) {
  return (float)__builtin_bit_cast(__bf16, s);
}
static __device__ __forceinline__ v8f zero8() {
  v8f z = {0.f, 0.f, 0.f, 0.f, 0.f, 0.f, 0.f, 0.f};
  return z;
}
// sum of 8 contiguous bf16 at 16B-aligned p, times-accumulated with w[8]
static __device__ __forceinline__ float dot8_bf(const unsigned short* p,
                                                const float* w) {
  union { uint4 q; unsigned short s[8]; } u;
  u.q = *reinterpret_cast<const uint4*>(p);
  float acc = 0.f;
#pragma unroll
  for (int i = 0; i < 8; ++i) acc += bf2f(u.s[i]) * w[i];
  return acc;
}
static __device__ __forceinline__ float sum8_bf(const unsigned short* p) {
  union { uint4 q; unsigned short s[8]; } u;
  u.q = *reinterpret_cast<const uint4*>(p);
  float acc = 0.f;
#pragma unroll
  for (int i = 0; i < 8; ++i) acc += bf2f(u.s[i]);
  return acc;
}

// A fragment: 16x32 bf16, row-major source (element (m,k) at base[m*ld + k]).
static __device__ __forceinline__ v16bf loadA(const unsigned short* base, int ld,
                                              int row0, int k0, int lane) {
  int r  = row0 + (lane & 15);
  int kb = k0 + ((lane >> 4) << 3);
  union { v16bf v; uint4 q[2]; } u;
  const unsigned short* p = base + (size_t)r * ld + kb;
  u.q[0] = *reinterpret_cast<const uint4*>(p);
  u.q[1] = *reinterpret_cast<const uint4*>(p + 16);
  return u.v;
}

// B fragment: 32x16 bf16 from an N-MAJOR source (element (k,n) at base[n*ld+k]).
static __device__ __forceinline__ v16bf loadBn(const unsigned short* base, int ld,
                                               int k0, int col0, int lane) {
  int n  = col0 + (lane & 15);
  int kb = k0 + ((lane >> 4) << 3);
  union { v16bf v; uint4 q[2]; } u;
  const unsigned short* p = base + (size_t)n * ld + kb;
  u.q[0] = *reinterpret_cast<const uint4*>(p);
  u.q[1] = *reinterpret_cast<const uint4*>(p + 16);
  return u.v;
}

static __device__ __forceinline__ v8f wmma_bf16(v16bf a, v16bf b, v8f c) {
  return __builtin_amdgcn_wmma_f32_16x16x32_bf16(false, a, false, b, (short)0, c,
                                                 false, false);
}

// CDNA5 async global->LDS b128 copy (ASYNCcnt-tracked, CDNA5 ISA)
static __device__ __forceinline__ void async_b128(unsigned lds_off,
                                                  const void* gaddr) {
  asm volatile("global_load_async_to_lds_b128 %0, %1, off"
               :: "v"(lds_off), "v"((unsigned long long)gaddr) : "memory");
}
static __device__ __forceinline__ void wait_async0() {
  asm volatile("s_wait_asynccnt 0x0" ::: "memory");
}

// ------------- kernel 1: weights fp32 -> bf16, TRANSPOSED ---------------
__global__ void cvt_w(const float* __restrict__ Wq, const float* __restrict__ Wk,
                      const float* __restrict__ Wv, unsigned short* __restrict__ Wt) {
  int idx = blockIdx.x * blockDim.x + threadIdx.x;
  const int per = DMODEL * DHEAD;
  int y = idx / per, rem = idx % per;
  int n = rem >> 10, k = rem & (DMODEL - 1);
  const float* src = (y == 0) ? Wq : (y == 1 ? Wk : Wv);
  Wt[idx] = f2bf(src[(size_t)k * DHEAD + n]);
}

// -------- kernel 2: fused QKV projection + feature map (elu+1) ----------
__global__ __launch_bounds__(256) void qkv_proj(
    const float* __restrict__ x, const unsigned short* __restrict__ Wt,
    unsigned short* __restrict__ Qws, unsigned short* __restrict__ Ktws,
    unsigned short* __restrict__ Knws, unsigned short* __restrict__ Vtws) {
  __shared__ alignas(16) unsigned short xbf[64 * 40];    //  5 KB
  __shared__ alignas(16) unsigned short wbf[128 * 40];   // 10 KB (n-major tile)
  const int tid = threadIdx.x, lane = tid & 31, w = tid >> 5;
  const int rowBase = blockIdx.x * 64;
  const int y = blockIdx.y;
  const unsigned short* Wsrc = Wt + (size_t)y * DMODEL * DHEAD;
  const int i = w & 3, jb = (w >> 2) * 4;
  v8f o[4];
#pragma unroll
  for (int jj = 0; jj < 4; ++jj) o[jj] = zero8();

  const int nn = tid >> 1;                 // W-tile row (n) handled by this thread
  const int wc = (tid & 1) * 16;           // W-tile k-offset
  const unsigned wlds = (unsigned)(unsigned long long)(wbf + nn * 40 + wc);

  for (int k0 = 0; k0 < DMODEL; k0 += 32) {
    // async-stage Wt tile (128 n x 32 k): 2 x b128 per thread, via TDM-path
    {
      const unsigned short* gp = Wsrc + (size_t)nn * DMODEL + k0 + wc;
      async_b128(wlds, gp);
      async_b128(wlds + 16, gp + 8);
    }
    {   // stage x tile (64 rows x 32 k) as bf16: 8 floats/thread
      int rr = tid >> 2;
      int cc = (tid & 3) * 8;
      const float* px = x + (size_t)(rowBase + rr) * DMODEL + k0 + cc;
      float4 a0 = reinterpret_cast<const float4*>(px)[0];
      float4 a1 = reinterpret_cast<const float4*>(px)[1];
      unsigned short* d = xbf + rr * 40 + cc;
      d[0] = f2bf(a0.x); d[1] = f2bf(a0.y); d[2] = f2bf(a0.z); d[3] = f2bf(a0.w);
      d[4] = f2bf(a1.x); d[5] = f2bf(a1.y); d[6] = f2bf(a1.z); d[7] = f2bf(a1.w);
    }
    wait_async0();
    __syncthreads();
    v16bf a = loadA(xbf, 40, 16 * i, 0, lane);
#pragma unroll
    for (int jj = 0; jj < 4; ++jj) {
      v16bf bb = loadBn(wbf, 40, 0, 16 * (jb + jj), lane);
      o[jj] = wmma_bf16(a, bb, o[jj]);
    }
    __syncthreads();
  }

  const int laneHi = lane >> 4, ln = lane & 15;
  const int tb   = rowBase + 16 * i + 8 * laneHi;      // first time row (8 contig)
  const int bb   = tb >> 12;
  const int tloc = tb & (SEQT - 1);
#pragma unroll
  for (int jj = 0; jj < 4; ++jj) {
    int col = 16 * (jb + jj) + ln;                     // output feature
    if (y == 0) {          // phi(q): row-major [t][feat] (A-operand use)
#pragma unroll
      for (int r = 0; r < 8; ++r) {
        float vv = o[jj][r];
        vv = vv > 0.f ? vv + 1.f : __expf(vv);
        Qws[(size_t)(tb + r) * DHEAD + col] = f2bf(vv);
      }
    } else if (y == 1) {   // phi(k): BOTH [feat][t] and [t][feat]
      union { uint4 q; unsigned short s[8]; } pk;
#pragma unroll
      for (int r = 0; r < 8; ++r) {
        float vv = o[jj][r];
        vv = vv > 0.f ? vv + 1.f : __expf(vv);
        pk.s[r] = f2bf(vv);
        Knws[(size_t)(bb * SEQT + tloc + r) * DHEAD + col] = pk.s[r];
      }
      *reinterpret_cast<uint4*>(Ktws + ((size_t)bb * DHEAD + col) * SEQT + tloc) = pk.q;
    } else {               // v: transposed [vdim][t] (B-operand uses)
      union { uint4 q; unsigned short s[8]; } pk;
#pragma unroll
      for (int r = 0; r < 8; ++r) pk.s[r] = f2bf(o[jj][r]);
      *reinterpret_cast<uint4*>(Vtws + ((size_t)bb * DHEAD + col) * SEQT + tloc) = pk.q;
    }
  }
}

// ---- kernel 3: per-partition partial state KV_p = phi_k^T V, ks_p ------
__global__ __launch_bounds__(256) void kv_partial(
    const unsigned short* __restrict__ Ktws, const unsigned short* __restrict__ Vtws,
    float* __restrict__ KVpart, float* __restrict__ KSpart) {
  const int p = blockIdx.x, b = blockIdx.y;
  const int tid = threadIdx.x, lane = tid & 31, w = tid >> 5;
  const unsigned short* KtB = Ktws + (size_t)b * DHEAD * SEQT;
  const unsigned short* VtB = Vtws + (size_t)b * DHEAD * SEQT;
  v8f acc[8];
#pragma unroll
  for (int j = 0; j < 8; ++j) acc[j] = zero8();
  for (int st = 0; st < TPART / 32; ++st) {
    int k0 = p * TPART + 32 * st;                 // time as K-dim
    v16bf a = loadA(KtB, SEQT, 16 * w, k0, lane); // A = K^T (feat x time)
#pragma unroll
    for (int j = 0; j < 8; ++j) {
      v16bf bb = loadBn(VtB, SEQT, k0, 16 * j, lane);
      acc[j] = wmma_bf16(a, bb, acc[j]);
    }
  }
  const int laneHi = lane >> 4, ln = lane & 15;
  float* dst = KVpart + (size_t)(b * NPART + p) * DHEAD * DHEAD;
#pragma unroll
  for (int j = 0; j < 8; ++j)
#pragma unroll
    for (int r = 0; r < 8; ++r) {
      int row = 16 * w + r + 8 * laneHi;
      dst[(size_t)row * DHEAD + 16 * j + ln] = acc[j][r];
    }
  if (tid < DHEAD) {
    float s = 0.f;
    const unsigned short* kr = KtB + (size_t)tid * SEQT + p * TPART;
    for (int t = 0; t < TPART; t += 8) s += sum8_bf(kr + t);
    KSpart[(size_t)(b * NPART + p) * DHEAD + tid] = s;
  }
}

// -------- kernel 4: chunked causal linear attention (per b, partition) --
__global__ __launch_bounds__(256) void lin_attn(
    const unsigned short* __restrict__ Qws, const unsigned short* __restrict__ Ktws,
    const unsigned short* __restrict__ Knws, const unsigned short* __restrict__ Vtws,
    const float* __restrict__ KVpart, const float* __restrict__ KSpart,
    float* __restrict__ out) {
  __shared__ alignas(16) unsigned short kvT_bf[128 * 128]; // [vdim][feat], 32 KB
  __shared__ alignas(16) unsigned short s_bf[64 * 64];     //  8 KB
  __shared__ float den[64];
  __shared__ float ks[128];
  const int p = blockIdx.x, b = blockIdx.y;
  const int tid = threadIdx.x, lane = tid & 31, w = tid >> 5;
  const int laneHi = lane >> 4, ln = lane & 15;
  const unsigned short* Qb  = Qws + (size_t)b * SEQT * DHEAD;
  const unsigned short* KtB = Ktws + (size_t)b * DHEAD * SEQT;
  const unsigned short* KnB = Knws + (size_t)b * SEQT * DHEAD;
  const unsigned short* VtB = Vtws + (size_t)b * DHEAD * SEQT;
  float* outB = out + (size_t)b * SEQT * DHEAD;

  // init running state from prefix of partition partials (sequence-parallel)
  v8f kvacc[8];     // wave w owns KV feature rows [16w,16w+16), fp32 master
#pragma unroll
  for (int j = 0; j < 8; ++j) kvacc[j] = zero8();
  for (int pp = 0; pp < p; ++pp) {
    const float* src = KVpart + (size_t)(b * NPART + pp) * DHEAD * DHEAD;
#pragma unroll
    for (int j = 0; j < 8; ++j)
#pragma unroll
      for (int r = 0; r < 8; ++r) {
        int row = 16 * w + r + 8 * laneHi;
        kvacc[j][r] += src[(size_t)row * DHEAD + 16 * j + ln];
      }
  }
  if (tid < DHEAD) {
    float s = 0.f;
    for (int pp = 0; pp < p; ++pp)
      s += KSpart[(size_t)(b * NPART + pp) * DHEAD + tid];
    ks[tid] = s;
  }
  __syncthreads();

  for (int c = 0; c < TPART / CHUNK; ++c) {
    const int t0 = p * TPART + c * CHUNK;

    // prefetch next chunk's operand rows into cache while this chunk computes
    if (c + 1 < TPART / CHUNK) {
      const int tn = t0 + CHUNK;
      // 64 Q rows + 64 Kn rows (256B each): thread -> quarter-row
      {
        int rr = tid >> 2, seg = (tid & 3) * 32;
        __builtin_prefetch(Qb + (size_t)(tn + rr) * DHEAD + seg, 0, 3);
        __builtin_prefetch(KnB + (size_t)(tn + rr) * DHEAD + seg, 0, 3);
      }
      // Vt / Kt: 128 feature rows x 64 new t columns (128B each)
      {
        int fr = tid >> 1, seg = (tid & 1) * 32;
        __builtin_prefetch(VtB + (size_t)fr * SEQT + tn + seg, 0, 3);
        __builtin_prefetch(KtB + (size_t)fr * SEQT + tn + seg, 0, 3);
      }
    }

    // (a) publish KV state transposed as bf16: kvT[vdim][feat]; one b128/tile/lane
#pragma unroll
    for (int j = 0; j < 8; ++j) {
      union { uint4 q; unsigned short s[8]; } pk;
#pragma unroll
      for (int r = 0; r < 8; ++r) pk.s[r] = f2bf(kvacc[j][r]);
      *reinterpret_cast<uint4*>(
          kvT_bf + (16 * j + ln) * 128 + 16 * w + 8 * laneHi) = pk.q;
    }
    if (tid < 64) den[tid] = 0.f;
    __syncthreads();

    // (b) intra-chunk scores S = tril(Q K^T); 16 tiles, 2 per wave
#pragma unroll
    for (int st = 0; st < 2; ++st) {
      int idx = 2 * w + st, i = idx >> 2, jj = idx & 3;
      if (jj <= i) {
        v8f sc = zero8();
#pragma unroll
        for (int kk = 0; kk < 4; ++kk) {          // feature dim 128
          v16bf a  = loadA(Qb, DHEAD, t0 + 16 * i, 32 * kk, lane);
          v16bf bb = loadBn(KnB, DHEAD, 32 * kk, t0 + 16 * jj, lane);
          sc = wmma_bf16(a, bb, sc);
        }
#pragma unroll
        for (int r = 0; r < 8; ++r) {
          int ri = r + 8 * laneHi;
          float v = sc[r];
          if (jj == i && ln > ri) v = 0.f;        // causal mask
          float rs = v;                           // row-sum -> denominator
#pragma unroll
          for (int m = 1; m < 16; m <<= 1) rs += __shfl_xor(rs, m, 16);
          if (ln == 0) atomicAdd(&den[16 * i + ri], rs);
          s_bf[(16 * i + ri) * 64 + 16 * jj + ln] = f2bf(v);
        }
      } else {
#pragma unroll
        for (int r = 0; r < 8; ++r)
          s_bf[(16 * i + r + 8 * laneHi) * 64 + 16 * jj + ln] = 0;
      }
    }
    // (c) inter-chunk denominator: q . ks (reads ks before the update below)
    if (tid < CHUNK) {
      const unsigned short* qr = Qb + (size_t)(t0 + tid) * DHEAD;
      float s = 0.f;
#pragma unroll
      for (int f = 0; f < DHEAD; f += 8) s += dot8_bf(qr + f, &ks[f]);
      atomicAdd(&den[tid], s);
    }
    __syncthreads();

    // (d) O = S_bf @ V + Q @ KV ; divide by den ; store fp32
    {
      const int i = w & 3, jb = (w >> 2) * 4;
      v8f o[4];
#pragma unroll
      for (int jj = 0; jj < 4; ++jj) o[jj] = zero8();
      for (int kk = 0; kk < 4; ++kk) {            // Q @ KV (feature K=128)
        v16bf a = loadA(Qb, DHEAD, t0 + 16 * i, 32 * kk, lane);
#pragma unroll
        for (int jj = 0; jj < 4; ++jj) {
          v16bf bb = loadBn(kvT_bf, 128, 32 * kk, 16 * (jb + jj), lane);
          o[jj] = wmma_bf16(a, bb, o[jj]);
        }
      }
#pragma unroll
      for (int kk = 0; kk < 2; ++kk) {            // S @ V (chunk K=64)
        v16bf a = loadA(s_bf, 64, 16 * i, 32 * kk, lane);
#pragma unroll
        for (int jj = 0; jj < 4; ++jj) {
          v16bf bb = loadBn(VtB, SEQT, t0 + 32 * kk, 16 * (jb + jj), lane);
          o[jj] = wmma_bf16(a, bb, o[jj]);
        }
      }
#pragma unroll
      for (int jj = 0; jj < 4; ++jj)
#pragma unroll
        for (int r = 0; r < 8; ++r) {
          int rl = 16 * i + r + 8 * laneHi;
          outB[(size_t)(t0 + rl) * DHEAD + 16 * (jb + jj) + ln] =
              o[jj][r] / den[rl];
        }
    }
    // (e) state update: KV += K^T V (fp32 accum); ks += colsum(phi_k)
#pragma unroll
    for (int st = 0; st < 2; ++st) {
      v16bf a = loadA(KtB, SEQT, 16 * w, t0 + 32 * st, lane);
#pragma unroll
      for (int j = 0; j < 8; ++j) {
        v16bf bb = loadBn(VtB, SEQT, t0 + 32 * st, 16 * j, lane);
        kvacc[j] = wmma_bf16(a, bb, kvacc[j]);
      }
    }
    if (tid < DHEAD) {
      const unsigned short* kr = KtB + (size_t)tid * SEQT + t0;
      float s = 0.f;
#pragma unroll
      for (int t = 0; t < CHUNK; t += 8) s += sum8_bf(kr + t);
      ks[tid] += s;
    }
    __syncthreads();
  }
}

extern "C" void kernel_launch(void* const* d_in, const int* in_sizes, int n_in,
                              void* d_out, int out_size, void* d_ws, size_t ws_size,
                              hipStream_t stream) {
  (void)in_sizes; (void)n_in; (void)out_size; (void)ws_size;
  const float* x  = (const float*)d_in[0];
  const float* Wq = (const float*)d_in[1];
  const float* Wk = (const float*)d_in[2];
  const float* Wv = (const float*)d_in[3];
  float* out = (float*)d_out;

  char* ws = (char*)d_ws;
  unsigned short* Wt   = (unsigned short*)(ws);                          // 768 KB
  unsigned short* Qws  = (unsigned short*)(ws + 786432);                 // 4 MB
  unsigned short* Ktws = (unsigned short*)(ws + 786432 + 1 * 4194304);   // 4 MB
  unsigned short* Knws = (unsigned short*)(ws + 786432 + 2 * 4194304);   // 4 MB
  unsigned short* Vtws = (unsigned short*)(ws + 786432 + 3 * 4194304);   // 4 MB
  float* KVpart = (float*)(ws + 786432 + 4 * 4194304);                   // 2 MB
  float* KSpart = (float*)(ws + 786432 + 4 * 4194304 + 2097152);         // 16 KB

  cvt_w<<<(3 * DMODEL * DHEAD) / 256, 256, 0, stream>>>(Wq, Wk, Wv, Wt);
  qkv_proj<<<dim3((BATCH * SEQT) / 64, 3), 256, 0, stream>>>(x, Wt, Qws, Ktws,
                                                             Knws, Vtws);
  kv_partial<<<dim3(NPART, BATCH), 256, 0, stream>>>(Ktws, Vtws, KVpart, KSpart);
  lin_attn<<<dim3(NPART, BATCH), 256, 0, stream>>>(Qws, Ktws, Knws, Vtws,
                                                   KVpart, KSpart, out);
}
